// WanTransformerBlock_VSA_78503412236470
// MI455X (gfx1250) — matrix operand order repaired and checked by reading, not compiled, not run on hardware
//
#include <hip/hip_runtime.h>
#include <hip/hip_bf16.h>

// ---------------- problem constants ----------------
#define S_LEN   4096
#define DMODEL  1536
#define NHEADS  12
#define DHEAD   128
#define LENC    512
#define FFDIM   8960
#define NBLK    64      // S_LEN / 64
#define TOPK    8
#define EPS_LN  1e-6f

typedef __bf16 bf16;
typedef __bf16 v16bf  __attribute__((ext_vector_type(16)));
typedef __bf16 bf16x8 __attribute__((ext_vector_type(8)));
typedef float  v8f    __attribute__((ext_vector_type(8)));

// ---------------- WMMA fragment loaders (contiguous 2 x 16B) ----------------
// A fragment (16x32 bf16): element j of the low 8 maps to K = half*8 + j,
// high 8 to K = 16 + half*8 + j.   p = &row[k_base]
__device__ __forceinline__ v16bf load_a_frag(const bf16* p, int half) {
    v16bf a;
    *(bf16x8*)&a       = *(const bf16x8*)(p + half * 8);
    *((bf16x8*)&a + 1) = *(const bf16x8*)(p + 16 + half * 8);
    return a;
}
// B fragment (32x16 bf16, operand stored K-contiguous per output column):
// element j of the low 8 maps to K = half*8 + j, high 8 to K = 16 + half*8 + j.   p = &col_row[k_base]
__device__ __forceinline__ v16bf load_b_frag(const bf16* p, int half) {
    v16bf b;
    *(bf16x8*)&b       = *(const bf16x8*)(p + half * 8);
    *((bf16x8*)&b + 1) = *(const bf16x8*)(p + 16 + half * 8);
    return b;
}

// branch-free tanh-approx GELU: tanh(u) = 1 - 2/(1+exp(2u))
__device__ __forceinline__ float gelu_tanh(float x) {
    float u = 0.7978845608028654f * (x + 0.044715f * x * x * x);
    float t = 1.0f - 2.0f / (1.0f + __expf(2.0f * u));
    return 0.5f * x * (1.0f + t);
}

// ---------------- small helpers ----------------
__device__ __forceinline__ float block_reduce_sum256(float v, float* sbuf) {
    int tid = threadIdx.x;
    sbuf[tid] = v;
    __syncthreads();
    for (int s = 128; s > 0; s >>= 1) {
        if (tid < s) sbuf[tid] += sbuf[tid + s];
        __syncthreads();
    }
    float r = sbuf[0];
    __syncthreads();
    return r;
}

// ---------------- elementwise kernels ----------------
__global__ void addvec_kernel(const float* __restrict__ a, const float* __restrict__ b,
                              float* __restrict__ o, int n) {
    int i = blockIdx.x * blockDim.x + threadIdx.x;
    if (i < n) o[i] = a[i] + b[i];
}

__global__ void f2b_kernel(const float* __restrict__ x, bf16* __restrict__ y, int n) {
    int i = blockIdx.x * blockDim.x + threadIdx.x;
    if (i < n) y[i] = (bf16)x[i];
}

// fp32 [M][N] -> bf16 transposed [N][M], LDS-tiled 32x32
__global__ __launch_bounds__(256)
void f2bT_kernel(const float* __restrict__ x, bf16* __restrict__ y, int M, int N) {
    __shared__ float tile[32][33];
    int bx = blockIdx.x * 32;          // col base (N)
    int by = blockIdx.y * 32;          // row base (M)
    int tx = threadIdx.x & 31, ty = threadIdx.x >> 5;
    for (int i = ty; i < 32; i += 8)
        tile[i][tx] = x[(size_t)(by + i) * N + bx + tx];
    __syncthreads();
    for (int i = ty; i < 32; i += 8)
        y[(size_t)(bx + i) * M + by + tx] = (bf16)tile[tx][i];
}

// out = a + b * (gate ? gate[ch] : 1)
__global__ void resid_kernel(const float* __restrict__ a, const float* __restrict__ b,
                             const float* __restrict__ gate, float* __restrict__ o, int n) {
    int i = blockIdx.x * blockDim.x + threadIdx.x;
    if (i < n) {
        float gv = gate ? gate[i % DMODEL] : 1.0f;
        o[i] = a[i] + b[i] * gv;
    }
}

// ---------------- LayerNorm (affine, optional 1+gamma), bf16 output ----------------
__global__ __launch_bounds__(256)
void ln_kernel(const float* __restrict__ x, const float* __restrict__ gamma,
               const float* __restrict__ beta, int addOne, bf16* __restrict__ out) {
    __shared__ float red[256];
    int row = blockIdx.x, tid = threadIdx.x;
    const float* xr = x + (size_t)row * DMODEL;
    float s = 0.f;
    for (int d = tid; d < DMODEL; d += 256) s += xr[d];
    s = block_reduce_sum256(s, red);
    float mean = s * (1.0f / DMODEL);
    float vs = 0.f;
    for (int d = tid; d < DMODEL; d += 256) { float t = xr[d] - mean; vs += t * t; }
    vs = block_reduce_sum256(vs, red);
    float rstd = rsqrtf(vs * (1.0f / DMODEL) + EPS_LN);
    bf16* orow = out + (size_t)row * DMODEL;
    float add1 = addOne ? 1.0f : 0.0f;
    for (int d = tid; d < DMODEL; d += 256) {
        float gm = gamma[d] + add1;
        orow[d] = (bf16)((xr[d] - mean) * rstd * gm + beta[d]);
    }
}

// ---------------- RMSNorm (+optional fused RoPE), bf16 output ----------------
__global__ __launch_bounds__(256)
void rms_kernel(const float* __restrict__ x, const float* __restrict__ w,
                const float* __restrict__ cosb, const float* __restrict__ sinb,
                bf16* __restrict__ out, int do_rope) {
    __shared__ float red[256];
    int row = blockIdx.x, tid = threadIdx.x;
    const float* xr = x + (size_t)row * DMODEL;
    float s = 0.f;
    for (int d = tid; d < DMODEL; d += 256) { float t = xr[d]; s += t * t; }
    s = block_reduce_sum256(s, red);
    float rstd = rsqrtf(s * (1.0f / DMODEL) + EPS_LN);
    bf16* orow = out + (size_t)row * DMODEL;
    if (do_rope) {
        for (int pr = tid; pr < DMODEL / 2; pr += 256) {
            int d0 = 2 * pr, d1 = d0 + 1;
            float x1 = xr[d0] * rstd * w[d0];
            float x2 = xr[d1] * rstd * w[d1];
            int pi = (d0 & (DHEAD - 1)) >> 1;
            float c  = cosb[(size_t)row * (DHEAD / 2) + pi];
            float sn = sinb[(size_t)row * (DHEAD / 2) + pi];
            orow[d0] = (bf16)(x1 * c - x2 * sn);
            orow[d1] = (bf16)(x2 * c + x1 * sn);
        }
    } else {
        for (int d = tid; d < DMODEL; d += 256)
            orow[d] = (bf16)(xr[d] * rstd * w[d]);
    }
}

// ---------------- bf16 WMMA GEMM: C[M,N] = A[M,K] @ W[K,N] + bias ----------------
// B operand pre-transposed: BT[N][K].  128x128 block tile, BK=32, double-buffered
// LDS, 8 waves (2x4), each wave 64x32 = 4x2 wmma tiles, all operands 2 x b128.
__global__ __launch_bounds__(256)
void gemm_bf16_kernel(const bf16* __restrict__ A, const bf16* __restrict__ BT,
                      const float* __restrict__ bias,
                      float* __restrict__ Cf, bf16* __restrict__ Cb,
                      int M, int N, int K, int gelu) {
    __shared__ bf16 As[2][128][32];
    __shared__ bf16 Bs[2][128][32];
    const int tid  = threadIdx.x;
    const int wave = tid >> 5, lane = tid & 31;
    const int l16  = lane & 15, half = lane >> 4;
    const int wm   = wave >> 2;      // 0..1
    const int wn   = wave & 3;       // 0..3
    const int bm   = blockIdx.y * 128;
    const int bn   = blockIdx.x * 128;

    // chunk mapping (identical for A and BT tiles): 512 chunks of 8 bf16
    const int cidx0 = tid,        r0 = cidx0 >> 2, c0 = (cidx0 & 3) << 3;
    const int cidx1 = tid + 256,  r1 = cidx1 >> 2, c1 = (cidx1 & 3) << 3;
    const bf16* Ap0 = A  + (size_t)(bm + r0) * K + c0;
    const bf16* Ap1 = A  + (size_t)(bm + r1) * K + c1;
    const bf16* Bp0 = BT + (size_t)(bn + r0) * K + c0;
    const bf16* Bp1 = BT + (size_t)(bn + r1) * K + c1;

    v8f acc[4][2] = {};

    // preload tile 0
    *(bf16x8*)&As[0][r0][c0] = *(const bf16x8*)Ap0;
    *(bf16x8*)&As[0][r1][c1] = *(const bf16x8*)Ap1;
    *(bf16x8*)&Bs[0][r0][c0] = *(const bf16x8*)Bp0;
    *(bf16x8*)&Bs[0][r1][c1] = *(const bf16x8*)Bp1;
    __syncthreads();

    int cur = 0;
    for (int k0 = 0; k0 < K; k0 += 32) {
        const bool hn = (k0 + 32) < K;
        bf16x8 ra0, ra1, rb0, rb1;
        if (hn) {                              // stage next tile into registers
            ra0 = *(const bf16x8*)(Ap0 + k0 + 32);
            ra1 = *(const bf16x8*)(Ap1 + k0 + 32);
            rb0 = *(const bf16x8*)(Bp0 + k0 + 32);
            rb1 = *(const bf16x8*)(Bp1 + k0 + 32);
        }
        // compute on current buffer
        v16bf bfr[2];
#pragma unroll
        for (int nt = 0; nt < 2; ++nt)
            bfr[nt] = load_b_frag(&Bs[cur][wn * 32 + nt * 16 + l16][0], half);
#pragma unroll
        for (int mt = 0; mt < 4; ++mt) {
            v16bf a = load_a_frag(&As[cur][wm * 64 + mt * 16 + l16][0], half);
#pragma unroll
            for (int nt = 0; nt < 2; ++nt)
                acc[mt][nt] = __builtin_amdgcn_wmma_f32_16x16x32_bf16(
                    false, a, false, bfr[nt], (short)0, acc[mt][nt], false, false);
        }
        if (hn) {
            int nxt = cur ^ 1;
            *(bf16x8*)&As[nxt][r0][c0] = ra0;
            *(bf16x8*)&As[nxt][r1][c1] = ra1;
            *(bf16x8*)&Bs[nxt][r0][c0] = rb0;
            *(bf16x8*)&Bs[nxt][r1][c1] = rb1;
            __syncthreads();
            cur = nxt;
        }
    }

#pragma unroll
    for (int mt = 0; mt < 4; ++mt) {
#pragma unroll
        for (int nt = 0; nt < 2; ++nt) {
            int n = bn + wn * 32 + nt * 16 + l16;
            float bv = bias ? bias[n] : 0.0f;
#pragma unroll
            for (int v = 0; v < 8; ++v) {
                int m = bm + wm * 64 + mt * 16 + v + half * 8;
                float val = acc[mt][nt][v] + bv;
                if (gelu) val = gelu_tanh(val);
                size_t o = (size_t)m * N + n;
                if (Cf) Cf[o] = val;
                if (Cb) Cb[o] = (bf16)val;
            }
        }
    }
}

// ---------------- block-mean pooling (V is transposed [D][S]) ----------------
__global__ __launch_bounds__(128)
void pool_kernel(const bf16* __restrict__ q, const bf16* __restrict__ k,
                 const bf16* __restrict__ vt,
                 float* __restrict__ qc, float* __restrict__ kc, float* __restrict__ vc) {
    int hb = blockIdx.x;                 // h*NBLK + i
    int h = hb / NBLK, i = hb % NBLK;
    int d = threadIdx.x;
    float sq = 0.f, sk = 0.f, sv = 0.f;
    const bf16* vrow = vt + (size_t)(h * DHEAD + d) * S_LEN + i * 64;
    for (int t = 0; t < 64; ++t) {
        size_t o = (size_t)(i * 64 + t) * DMODEL + h * DHEAD + d;
        sq += (float)q[o]; sk += (float)k[o]; sv += (float)vrow[t];
    }
    size_t oo = (size_t)hb * DHEAD + d;
    qc[oo] = sq * (1.0f / 64.0f);
    kc[oo] = sk * (1.0f / 64.0f);
    vc[oo] = sv * (1.0f / 64.0f);
}

// ---------------- coarse attention per head ----------------
__global__ __launch_bounds__(256)
void coarse_kernel(const float* __restrict__ qc, const float* __restrict__ kc,
                   const float* __restrict__ vc,
                   float* __restrict__ p, float* __restrict__ oc) {
    __shared__ float Sc[64][64];
    int h = blockIdx.x, tid = threadIdx.x;
    const float scale = 0.08838834764831845f;
    const float* qh = qc + (size_t)h * NBLK * DHEAD;
    const float* kh = kc + (size_t)h * NBLK * DHEAD;
    const float* vh = vc + (size_t)h * NBLK * DHEAD;
    for (int t = tid; t < 64 * 64; t += 256) {
        int r = t >> 6, c = t & 63;
        float s = 0.f;
        for (int d = 0; d < DHEAD; ++d) s += qh[r * DHEAD + d] * kh[c * DHEAD + d];
        Sc[r][c] = s * scale;
    }
    __syncthreads();
    if (tid < 64) {
        float mx = -1e30f;
        for (int c = 0; c < 64; ++c) mx = fmaxf(mx, Sc[tid][c]);
        float sum = 0.f;
        for (int c = 0; c < 64; ++c) { float e = __expf(Sc[tid][c] - mx); Sc[tid][c] = e; sum += e; }
        float inv = 1.0f / sum;
        for (int c = 0; c < 64; ++c) {
            float pv = Sc[tid][c] * inv;
            Sc[tid][c] = pv;
            p[((size_t)h * 64 + tid) * 64 + c] = pv;
        }
    }
    __syncthreads();
    for (int t = tid; t < 64 * DHEAD; t += 256) {
        int r = t >> 7, d = t & (DHEAD - 1);
        float s = 0.f;
        for (int j = 0; j < 64; ++j) s += Sc[r][j] * vh[j * DHEAD + d];
        oc[((size_t)h * 64 + r) * DHEAD + d] = s;
    }
}

// ---------------- top-8 block indices per (h, qblock) ----------------
__global__ void topk_kernel(const float* __restrict__ p, int* __restrict__ idx) {
    int r = blockIdx.x * blockDim.x + threadIdx.x;   // h*NBLK + i
    if (r >= NHEADS * NBLK) return;
    float row[64];
    for (int c = 0; c < 64; ++c) row[c] = p[(size_t)r * 64 + c];
    for (int t = 0; t < TOPK; ++t) {
        int best = 0; float bv = row[0];
        for (int c = 1; c < 64; ++c) if (row[c] > bv) { bv = row[c]; best = c; }
        idx[r * TOPK + t] = best;
        row[best] = -1e30f;
    }
}

// ---------------- flash attention over 8 KV blocks of 64 (VSA-fine & cross) ----------------
// V transposed: vt[(h*DHEAD + d) * vt_stride + token].
__global__ __launch_bounds__(256)
void attn_kernel(const bf16* __restrict__ Q, const bf16* __restrict__ Kg,
                 const bf16* __restrict__ Vt, int vt_stride,
                 const int* __restrict__ idx, const float* __restrict__ oc,
                 const float* __restrict__ g, bf16* __restrict__ out) {
    __shared__ bf16  Qs[64][DHEAD];
    __shared__ float Ss[64][64];
    __shared__ bf16  Ps[64][64];
    __shared__ float rmax[64], rsum[64], ralpha[64];

    const int i = blockIdx.x, h = blockIdx.y;
    const int tid = threadIdx.x;
    const int wave = tid >> 5, lane = tid & 31;
    const int l16 = lane & 15, half = lane >> 4;
    const float scale = 0.08838834764831845f;

#pragma unroll
    for (int it = 0; it < 4; ++it) {            // load Q tile 64x128 bf16
        int cidx = tid + it * 256;
        int r = cidx >> 4, c = (cidx & 15) << 3;
        *(bf16x8*)&Qs[r][c] = *(const bf16x8*)(Q + (size_t)(i * 64 + r) * DMODEL + h * DHEAD + c);
    }
    if (tid < 64) { rmax[tid] = -1e30f; rsum[tid] = 0.f; }
    __syncthreads();

    v8f o_acc[4] = {};
    const int smt  = wave & 3;            // score m-tile for this wave
    const int snt0 = (wave >> 2) * 2;     // score n-tiles snt0, snt0+1

    for (int jj = 0; jj < TOPK; ++jj) {
        int blk = idx ? idx[((size_t)h * NBLK + i) * TOPK + jj] : jj;

        // prefetch next selected block's K/V slices (data-dependent gather)
        if (jj + 1 < TOPK) {
            int nb = idx ? idx[((size_t)h * NBLK + i) * TOPK + jj + 1] : jj + 1;
            const bf16* pk = Kg + (size_t)nb * 64 * DMODEL + h * DHEAD
                           + (size_t)(tid >> 2) * DMODEL + (tid & 3) * 32;
            __builtin_prefetch(pk, 0, 3);
            const bf16* pv = Vt + (size_t)(h * DHEAD + (tid >> 1)) * vt_stride
                           + nb * 64 + (tid & 1) * 32;
            __builtin_prefetch(pv, 0, 3);
        }

        const bf16* Kbase = Kg + (size_t)blk * 64 * DMODEL + h * DHEAD;

        // ---- scores S = Q K^T * scale (64x64) ----
#pragma unroll
        for (int nt = 0; nt < 2; ++nt) {
            v8f sacc = {};
            int n = (snt0 + nt) * 16 + l16;
            const bf16* Krow = Kbase + (size_t)n * DMODEL;
#pragma unroll
            for (int ks = 0; ks < 4; ++ks) {
                v16bf a = load_a_frag(&Qs[smt * 16 + l16][ks * 32], half);
                v16bf b = load_b_frag(Krow + ks * 32, half);
                sacc = __builtin_amdgcn_wmma_f32_16x16x32_bf16(
                    false, a, false, b, (short)0, sacc, false, false);
            }
#pragma unroll
            for (int v = 0; v < 8; ++v) {
                int m = smt * 16 + v + half * 8;
                Ss[m][n] = sacc[v] * scale;
            }
        }
        __syncthreads();

        // ---- online softmax rows ----
        if (tid < 64) {
            float mo = rmax[tid];
            float mx = mo;
            for (int c = 0; c < 64; ++c) mx = fmaxf(mx, Ss[tid][c]);
            float alpha = __expf(mo - mx);
            float sum = 0.f;
            for (int c = 0; c < 64; ++c) {
                float pv = __expf(Ss[tid][c] - mx);
                Ps[tid][c] = (bf16)pv;
                sum += pv;
            }
            rsum[tid] = rsum[tid] * alpha + sum;
            rmax[tid] = mx;
            ralpha[tid] = alpha;
        }
        __syncthreads();

        // ---- rescale + O += P V (wave owns 16 output cols) ----
        int n = wave * 16 + l16;
        const bf16* Vrow = Vt + (size_t)(h * DHEAD + n) * vt_stride + blk * 64;
#pragma unroll
        for (int mt = 0; mt < 4; ++mt) {
#pragma unroll
            for (int v = 0; v < 8; ++v) {
                int m = mt * 16 + v + half * 8;
                o_acc[mt][v] *= ralpha[m];
            }
        }
#pragma unroll
        for (int ks = 0; ks < 2; ++ks) {
            v16bf b = load_b_frag(Vrow + ks * 32, half);
#pragma unroll
            for (int mt = 0; mt < 4; ++mt) {
                v16bf a = load_a_frag(&Ps[mt * 16 + l16][ks * 32], half);
                o_acc[mt] = __builtin_amdgcn_wmma_f32_16x16x32_bf16(
                    false, a, false, b, (short)0, o_acc[mt], false, false);
            }
        }
        __syncthreads();
    }

    // ---- epilogue (reciprocal instead of per-element divide) ----
    int n = wave * 16 + l16;
#pragma unroll
    for (int mt = 0; mt < 4; ++mt) {
#pragma unroll
        for (int v = 0; v < 8; ++v) {
            int m = mt * 16 + v + half * 8;
            float val = o_acc[mt][v] * __frcp_rn(rsum[m]);
            int token = i * 64 + m;
            if (oc) {
                float ocv = oc[((size_t)h * NBLK + i) * DHEAD + n];
                float gv  = g[(size_t)token * DMODEL + h * DHEAD + n];
                val += ocv * gv;
            }
            out[(size_t)token * DMODEL + h * DHEAD + n] = (bf16)val;
        }
    }
}

// ---------------- host orchestration ----------------
extern "C" void kernel_launch(void* const* d_in, const int* in_sizes, int n_in,
                              void* d_out, int out_size, void* d_ws, size_t ws_size,
                              hipStream_t stream) {
    const float* h_in  = (const float*)d_in[0];
    const float* enc   = (const float*)d_in[1];
    const float* temb  = (const float*)d_in[2];
    const float* cosb  = (const float*)d_in[3];
    const float* sinb  = (const float*)d_in[4];
    const float* sst   = (const float*)d_in[5];
    const float* Wq = (const float*)d_in[6],  *bq = (const float*)d_in[7];
    const float* Wk = (const float*)d_in[8],  *bk = (const float*)d_in[9];
    const float* Wv = (const float*)d_in[10], *bv = (const float*)d_in[11];
    const float* Wg = (const float*)d_in[12], *bg = (const float*)d_in[13];
    const float* Wo = (const float*)d_in[14], *bo = (const float*)d_in[15];
    const float* nq = (const float*)d_in[16], *nk = (const float*)d_in[17];
    const float* ln1w = (const float*)d_in[18], *ln1b = (const float*)d_in[19];
    const float* cWq = (const float*)d_in[20], *cbq = (const float*)d_in[21];
    const float* cWk = (const float*)d_in[22], *cbk = (const float*)d_in[23];
    const float* cWv = (const float*)d_in[24], *cbv = (const float*)d_in[25];
    const float* cWo = (const float*)d_in[26], *cbo = (const float*)d_in[27];
    const float* cnq = (const float*)d_in[28], *cnk = (const float*)d_in[29];
    const float* W1 = (const float*)d_in[30], *b1 = (const float*)d_in[31];
    const float* W2 = (const float*)d_in[32], *b2 = (const float*)d_in[33];
    // topk (d_in[34]) fixed at 8 by the reference; hardcoded as TOPK.

    const size_t szW  = (size_t)DMODEL * DMODEL;
    const size_t szW1 = (size_t)DMODEL * FFDIM;
    const size_t szSD = (size_t)S_LEN * DMODEL;
    const size_t szLD = (size_t)LENC * DMODEL;

    size_t off = 0;
    char* base = (char*)d_ws;
    auto alloc = [&](size_t bytes) -> void* {
        off = (off + 255) & ~(size_t)255;
        void* p = base + off;
        off += bytes;
        return p;
    };

    // transposed bf16 weights WT[N][K]
    bf16* wtWq = (bf16*)alloc(szW * 2);
    bf16* wtWk = (bf16*)alloc(szW * 2);
    bf16* wtWv = (bf16*)alloc(szW * 2);
    bf16* wtWg = (bf16*)alloc(szW * 2);
    bf16* wtWo = (bf16*)alloc(szW * 2);
    bf16* wtcWq = (bf16*)alloc(szW * 2);
    bf16* wtcWk = (bf16*)alloc(szW * 2);
    bf16* wtcWv = (bf16*)alloc(szW * 2);
    bf16* wtcWo = (bf16*)alloc(szW * 2);
    bf16* wtW1 = (bf16*)alloc(szW1 * 2);
    bf16* wtW2 = (bf16*)alloc(szW1 * 2);

    float* evec   = (float*)alloc(6 * DMODEL * 4);
    bf16*  encb   = (bf16*)alloc(szLD * 2);
    bf16*  norm1b = (bf16*)alloc(szSD * 2);
    float* qf = (float*)alloc(szSD * 4);
    float* kf = (float*)alloc(szSD * 4);
    float* vf = (float*)alloc(szSD * 4);
    float* gf = (float*)alloc(szSD * 4);
    bf16*  qb = (bf16*)alloc(szSD * 2);
    bf16*  kb = (bf16*)alloc(szSD * 2);
    bf16*  vtb = (bf16*)alloc(szSD * 2);      // V transposed [D][S]
    float* qc = (float*)alloc((size_t)NHEADS * NBLK * DHEAD * 4);
    float* kc = (float*)alloc((size_t)NHEADS * NBLK * DHEAD * 4);
    float* vc = (float*)alloc((size_t)NHEADS * NBLK * DHEAD * 4);
    float* pc = (float*)alloc((size_t)NHEADS * NBLK * NBLK * 4);
    float* occ = (float*)alloc((size_t)NHEADS * NBLK * DHEAD * 4);
    int*   idxb = (int*)alloc((size_t)NHEADS * NBLK * TOPK * 4);
    bf16*  attnb = (bf16*)alloc(szSD * 2);
    float* resid1 = (float*)alloc(szSD * 4);
    bf16*  norm2b = (bf16*)alloc(szSD * 2);
    bf16*  cqb = (bf16*)alloc(szSD * 2);
    float* ckf = (float*)alloc(szLD * 4);
    float* cvf = (float*)alloc(szLD * 4);
    bf16*  ckb = (bf16*)alloc(szLD * 2);
    bf16*  cvtb = (bf16*)alloc(szLD * 2);     // cross-V transposed [D][L]
    bf16*  cattnb = (bf16*)alloc(szSD * 2);
    float* resid2 = (float*)alloc(szSD * 4);
    bf16*  norm3b = (bf16*)alloc(szSD * 2);
    bf16*  ff1b = (bf16*)alloc((size_t)S_LEN * FFDIM * 2);
    // aliases of dead buffers:
    float* cqf   = qf;   // qf dead after rms+rope
    float* coutf = kf;   // kf dead after rms+rope
    float* ff2f  = vf;   // vf dead after transpose-convert
    float* attnO = gf;   // gf dead after VSA attention combine

    auto cvtT = [&](const float* src, bf16* dst, int M, int N) {
        f2bT_kernel<<<dim3(N / 32, M / 32), 256, 0, stream>>>(src, dst, M, N);
    };
    auto gemm = [&](const bf16* A, const bf16* BT, const float* bias,
                    float* Cf, bf16* Cb, int M, int N, int K, int gelu) {
        gemm_bf16_kernel<<<dim3(N / 128, M / 128), 256, 0, stream>>>(A, BT, bias, Cf, Cb, M, N, K, gelu);
    };

    // 0. modulation e = temb + sst
    addvec_kernel<<<dim3((6 * DMODEL + 255) / 256), 256, 0, stream>>>(temb, sst, evec, 6 * DMODEL);

    // 1. weights fp32 -> bf16, transposed to [N][K]
    cvtT(Wq, wtWq, DMODEL, DMODEL);  cvtT(Wk, wtWk, DMODEL, DMODEL);
    cvtT(Wv, wtWv, DMODEL, DMODEL);  cvtT(Wg, wtWg, DMODEL, DMODEL);
    cvtT(Wo, wtWo, DMODEL, DMODEL);
    cvtT(cWq, wtcWq, DMODEL, DMODEL); cvtT(cWk, wtcWk, DMODEL, DMODEL);
    cvtT(cWv, wtcWv, DMODEL, DMODEL); cvtT(cWo, wtcWo, DMODEL, DMODEL);
    cvtT(W1, wtW1, DMODEL, FFDIM);   cvtT(W2, wtW2, FFDIM, DMODEL);
    f2b_kernel<<<dim3((unsigned)((szLD + 255) / 256)), 256, 0, stream>>>(enc, encb, (int)szLD);

    // 2. norm1 = LN(h)*(1+scale)+shift
    ln_kernel<<<dim3(S_LEN), 256, 0, stream>>>(h_in, evec + DMODEL, evec, 1, norm1b);

    // 3. q,k,v,g projections
    gemm(norm1b, wtWq, bq, qf, nullptr, S_LEN, DMODEL, DMODEL, 0);
    gemm(norm1b, wtWk, bk, kf, nullptr, S_LEN, DMODEL, DMODEL, 0);
    gemm(norm1b, wtWv, bv, vf, nullptr, S_LEN, DMODEL, DMODEL, 0);
    gemm(norm1b, wtWg, bg, gf, nullptr, S_LEN, DMODEL, DMODEL, 0);

    // 4. RMS + RoPE -> bf16; V transposed for WMMA B operand
    rms_kernel<<<dim3(S_LEN), 256, 0, stream>>>(qf, nq, cosb, sinb, qb, 1);
    rms_kernel<<<dim3(S_LEN), 256, 0, stream>>>(kf, nk, cosb, sinb, kb, 1);
    cvtT(vf, vtb, S_LEN, DMODEL);

    // 5. VSA coarse path
    pool_kernel<<<dim3(NHEADS * NBLK), 128, 0, stream>>>(qb, kb, vtb, qc, kc, vc);
    coarse_kernel<<<dim3(NHEADS), 256, 0, stream>>>(qc, kc, vc, pc, occ);
    topk_kernel<<<dim3(3), 256, 0, stream>>>(pc, idxb);

    // 6. VSA fine attention (+ gated coarse)
    attn_kernel<<<dim3(NBLK, NHEADS), 256, 0, stream>>>(qb, kb, vtb, S_LEN, idxb, occ, gf, attnb);

    // 7. output projection + gated residual
    gemm(attnb, wtWo, bo, attnO, nullptr, S_LEN, DMODEL, DMODEL, 0);
    resid_kernel<<<dim3((unsigned)((szSD + 255) / 256)), 256, 0, stream>>>(
        h_in, attnO, evec + 2 * DMODEL, resid1, (int)szSD);

    // 8. norm2 + cross attention
    ln_kernel<<<dim3(S_LEN), 256, 0, stream>>>(resid1, ln1w, ln1b, 0, norm2b);
    gemm(norm2b, wtcWq, cbq, cqf, nullptr, S_LEN, DMODEL, DMODEL, 0);
    rms_kernel<<<dim3(S_LEN), 256, 0, stream>>>(cqf, cnq, nullptr, nullptr, cqb, 0);
    gemm(encb, wtcWk, cbk, ckf, nullptr, LENC, DMODEL, DMODEL, 0);
    rms_kernel<<<dim3(LENC), 256, 0, stream>>>(ckf, cnk, nullptr, nullptr, ckb, 0);
    gemm(encb, wtcWv, cbv, cvf, nullptr, LENC, DMODEL, DMODEL, 0);
    cvtT(cvf, cvtb, LENC, DMODEL);
    attn_kernel<<<dim3(NBLK, NHEADS), 256, 0, stream>>>(cqb, ckb, cvtb, LENC,
                                                        nullptr, nullptr, nullptr, cattnb);
    gemm(cattnb, wtcWo, cbo, coutf, nullptr, S_LEN, DMODEL, DMODEL, 0);
    resid_kernel<<<dim3((unsigned)((szSD + 255) / 256)), 256, 0, stream>>>(
        resid1, coutf, nullptr, resid2, (int)szSD);

    // 9. norm3 + FFN (GELU fused into first GEMM epilogue, bf16 out)
    ln_kernel<<<dim3(S_LEN), 256, 0, stream>>>(resid2, evec + 4 * DMODEL, evec + 3 * DMODEL, 1, norm3b);
    gemm(norm3b, wtW1, b1, nullptr, ff1b, S_LEN, FFDIM, DMODEL, 1);
    gemm(ff1b, wtW2, b2, ff2f, nullptr, S_LEN, DMODEL, FFDIM, 0);

    // 10. final gated residual -> output
    resid_kernel<<<dim3((unsigned)((szSD + 255) / 256)), 256, 0, stream>>>(
        resid2, ff2f, evec + 5 * DMODEL, (float*)d_out, (int)szSD);
}
